// LDLF_47708496724029
// MI455X (gfx1250) — matrix operand order repaired and checked by reading, compiled, not run on hardware
//
#include <hip/hip_runtime.h>
#include <hip/hip_bf16.h>

// ---------------------------------------------------------------------------
// Fused LDLF kernel for gfx1250 (CDNA5, wave32, WMMA bf16 16x16x32 + TDM)
//   h      = sigmoid(X @ W1 + b1)          (131072x256)@(256x384)
//   latent = sigmoid(h @ W2 + b2)          (131072x384)@(384x64)
//   mu     = tree-routing(latent[phi])     -> (N, 5, 64)
//   out    = (mu_flat @ pi_flat) / 5       (N x 320)@(320 x 16)
// One workgroup = 64 rows; W1 K-slices DMA'd by the Tensor Data Mover into
// LDS staging (TENSORcnt), then converted bf16; 3 WMMA GEMMs out of LDS.
// ---------------------------------------------------------------------------

typedef __bf16 bf16_t;
typedef __attribute__((ext_vector_type(16))) __bf16 v16bf;
typedef __attribute__((ext_vector_type(8)))  __bf16 v8bf;
typedef __attribute__((ext_vector_type(8)))  float  v8f;
typedef __attribute__((ext_vector_type(4)))  unsigned u32x4;
typedef __attribute__((ext_vector_type(8)))  int      i32x8;
typedef __attribute__((ext_vector_type(4)))  int      i32x4;

#define F_DIM   256
#define H_DIM   384
#define L_DIM   64
#define O_DIM   16
#define E_NUM   5
#define K_MU    320     // E_NUM * LEAVES
#define MT      64      // rows per workgroup
#define THREADS 256
#define W1_SLICE (32 * H_DIM)   // 12288 f32 per K-step slice

// LDS carve-out offsets (bytes); total 275456 < 320KB per WGP
#define OFF_SX    0        // bf16 [64][256]   32768
#define OFF_SW1T  32768    // bf16 [384][32]   24576  (n-major, k contiguous)
#define OFF_SH    57344    // bf16 [64][384]   49152
#define OFF_SW2T  106496   // bf16 [64][384]   49152  (n-major, k contiguous)
#define OFF_SLAT  155648   // f32  [64][64]    16384
#define OFF_SMU   172032   // bf16 [64][320]   40960
#define OFF_SPIT  212992   // bf16 [16][320]   10240  (o-major, k contiguous)
#define OFF_SPHI  223232   // i32  [320]        1280
#define OFF_SB1   224512   // f32  [384]        1536
#define OFF_SB2   226048   // f32  [64]          256
#define OFF_STAGE 226304   // f32  [12288]     49152  (TDM landing zone)
#define SMEM_BYTES 275456

__device__ __forceinline__ bf16_t f2bf(float f) {
    union { float f; unsigned u; } v; v.f = f;
    unsigned r = v.u + 0x7FFFu + ((v.u >> 16) & 1u);   // round-to-nearest-even
    unsigned short hs = (unsigned short)(r >> 16);
    bf16_t out;
    __builtin_memcpy(&out, &hs, 2);
    return out;
}

__device__ __forceinline__ float sigm(float x) {
    return 1.0f / (1.0f + __expf(-x));
}

__device__ __forceinline__ v16bf cat8(v8bf lo, v8bf hi) {
    return __builtin_shufflevector(lo, hi, 0,1,2,3,4,5,6,7,8,9,10,11,12,13,14,15);
}

__device__ __forceinline__ v8f zero8() {
    v8f z;
#pragma unroll
    for (int i = 0; i < 8; ++i) z[i] = 0.0f;
    return z;
}

// Tensor Data Mover: raw-copy nElems f32 (1 x nElems 2D tile) from global to
// LDS byte-offset ldsOff. D# layout per CDNA5 ISA ch.8 (group0/group1).
__device__ __forceinline__ void tdm_load_slice(unsigned ldsOff,
                                               const float* gptr,
                                               unsigned nElems) {
    unsigned long long ga = (unsigned long long)(const void*)gptr;
    u32x4 g0;
    g0[0] = 1u;                                          // count=1 (valid user D#)
    g0[1] = ldsOff;                                      // lds_addr (bytes)
    g0[2] = (unsigned)(ga & 0xFFFFFFFFu);                // global_addr[31:0]
    g0[3] = (unsigned)((ga >> 32) & 0x01FFFFFFu)         // global_addr[56:32]
          | (2u << 30);                                  // type=2 ("image")
    i32x8 g1;
    g1[0] = (int)(2u << 16);                             // data_size=4B
    g1[1] = (int)((nElems & 0xFFFFu) << 16);             // tensor_dim0[15:0]
    g1[2] = (int)(((nElems >> 16) & 0xFFFFu)             // tensor_dim0[31:16]
          | (1u << 16));                                 // tensor_dim1 = 1
    g1[3] = (int)((nElems & 0xFFFFu) << 16);             // tile_dim0 = nElems
    g1[4] = 1;                                           // tile_dim1=1, tile_dim2=0
    g1[5] = (int)nElems;                                 // tensor_dim0_stride lo32
    g1[6] = 0;                                           // stride hi / dim1_stride
    g1[7] = 0;
    i32x4 z4 = {0, 0, 0, 0};
#if defined(__clang_major__) && (__clang_major__ >= 23)
    i32x8 z8 = {0, 0, 0, 0, 0, 0, 0, 0};
    __builtin_amdgcn_tensor_load_to_lds(g0, g1, z4, z4, z8, 0);
#else
    __builtin_amdgcn_tensor_load_to_lds(g0, g1, z4, z4, 0);
#endif
}

__global__ void __launch_bounds__(THREADS)
ldlf_fused_kernel(const float* __restrict__ X,
                  const float* __restrict__ W1,
                  const float* __restrict__ b1,
                  const float* __restrict__ W2,
                  const float* __restrict__ b2,
                  const int*   __restrict__ phi,
                  const float* __restrict__ pi,
                  float*       __restrict__ out) {
    extern __shared__ char smem[];
    bf16_t* sX   = (bf16_t*)(smem + OFF_SX);
    bf16_t* sW1t = (bf16_t*)(smem + OFF_SW1T);
    bf16_t* sH   = (bf16_t*)(smem + OFF_SH);
    bf16_t* sW2t = (bf16_t*)(smem + OFF_SW2T);
    float*  sLat = (float*) (smem + OFF_SLAT);
    bf16_t* sMu  = (bf16_t*)(smem + OFF_SMU);
    bf16_t* sPiT = (bf16_t*)(smem + OFF_SPIT);
    int*    sPhi = (int*)   (smem + OFF_SPHI);
    float*  sB1  = (float*) (smem + OFF_SB1);
    float*  sB2  = (float*) (smem + OFF_SB2);
    float*  sStg = (float*) (smem + OFF_STAGE);

    const int tid  = threadIdx.x;
    const int lane = tid & 31;
    const int wave = tid >> 5;
    const int lm   = lane & 15;      // col-within-tile / M-row select
    const int lhi  = lane >> 4;      // 0: lanes 0-15, 1: lanes 16-31
    const int rowBase = blockIdx.x * MT;

    // -------- Prologue staging: X tile, W2^T, pi^T, phi, b1, b2 -----------
    {
        const float4* Xv = (const float4*)(X + (size_t)rowBase * F_DIM);
#pragma unroll
        for (int j = 0; j < 16; ++j) {                 // 64*256 floats
            int idx = j * THREADS + tid;
            float4 v = Xv[idx];
            int f0 = idx * 4;                          // row-major, same layout
            sX[f0 + 0] = f2bf(v.x); sX[f0 + 1] = f2bf(v.y);
            sX[f0 + 2] = f2bf(v.z); sX[f0 + 3] = f2bf(v.w);
        }
        const float4* W2v = (const float4*)W2;
#pragma unroll
        for (int j = 0; j < 24; ++j) {                 // 384*64 floats -> [n][k]
            int idx = j * THREADS + tid;
            float4 v = W2v[idx];
            int f0 = idx * 4;
            float vv[4] = {v.x, v.y, v.z, v.w};
#pragma unroll
            for (int q = 0; q < 4; ++q) {
                int f = f0 + q;
                int r = f >> 6;          // k (0..383)
                int c = f & 63;          // n (0..63)
                sW2t[c * H_DIM + r] = f2bf(vv[q]);
            }
        }
        const float4* Piv = (const float4*)pi;
#pragma unroll
        for (int j = 0; j < 5; ++j) {                  // 5*64*16 floats -> [o][k]
            int idx = j * THREADS + tid;
            float4 v = Piv[idx];
            int f0 = idx * 4;
            float vv[4] = {v.x, v.y, v.z, v.w};
#pragma unroll
            for (int q = 0; q < 4; ++q) {
                int f = f0 + q;
                int k = f >> 4;          // e*64+l (0..319)
                int o = f & 15;
                sPiT[o * K_MU + k] = f2bf(vv[q]);
            }
        }
        for (int i = tid; i < K_MU; i += THREADS) sPhi[i] = phi[i];
        for (int i = tid; i < H_DIM; i += THREADS) sB1[i] = b1[i];
        if (tid < L_DIM) sB2[tid] = b2[tid];
    }
    __syncthreads();

    // -------- Stage 1: h = sigmoid(X @ W1 + b1), WMMA bf16 ----------------
    const int rt = wave >> 1;       // row tile 0..3 (16 rows each)
    const int nh = wave & 1;        // N half: cols [nh*192, nh*192+192)

    v8f acc1[12];
#pragma unroll
    for (int t = 0; t < 12; ++t) acc1[t] = zero8();

    for (int kk = 0; kk < 8; ++kk) {                   // K = 256 in 8 x 32
        __syncthreads();   // prior sW1t reads + prior staging reads complete
        // --- TDM: DMA W1 rows [32kk, 32kk+32) f32 into LDS staging --------
        if (wave == 0) {
            tdm_load_slice(OFF_STAGE, W1 + (size_t)kk * W1_SLICE, W1_SLICE);
            __builtin_amdgcn_s_wait_tensorcnt(0);
        }
        if (kk < 7)                                    // warm L2 for next slice
            __builtin_prefetch(W1 + (size_t)(kk + 1) * W1_SLICE + tid * 4, 0, 1);
        __syncthreads();   // staging visible to all waves
        // --- convert staging f32 -> sW1t[n][k] bf16 (transposed) ----------
        const float4* Sv = (const float4*)sStg;
#pragma unroll
        for (int j = 0; j < 12; ++j) {                 // 32*384 floats
            int idx = j * THREADS + tid;
            float4 v = Sv[idx];
            int f0 = idx * 4;
            float vv[4] = {v.x, v.y, v.z, v.w};
#pragma unroll
            for (int q = 0; q < 4; ++q) {
                int f = f0 + q;
                int r = f / H_DIM;                     // k within slice (0..31)
                int c = f - r * H_DIM;                 // n (0..383)
                sW1t[c * 32 + r] = f2bf(vv[q]);
            }
        }
        __syncthreads();

        // A fragment: 16x32 bf16 from sX
        const int m  = rt * 16 + lm;
        const int k0 = kk * 32 + lhi * 16;
        v8bf alo = *(const v8bf*)&sX[m * F_DIM + k0];
        v8bf ahi = *(const v8bf*)&sX[m * F_DIM + k0 + 8];
        v16bf a = cat8(alo, ahi);

        // software-pipelined B fragments: keep next DS load in flight
        const int nBase = (nh * 12) * 16 + lm;
        v16bf bcur = *(const v16bf*)&sW1t[nBase * 32 + lhi * 16];
#pragma unroll
        for (int t = 0; t < 12; ++t) {
            v16bf bnext = bcur;
            if (t < 11)
                bnext = *(const v16bf*)&sW1t[(nBase + (t + 1) * 16) * 32 + lhi * 16];
            acc1[t] = __builtin_amdgcn_wmma_f32_16x16x32_bf16(
                false, a, false, bcur, (short)0, acc1[t], false, false);
            bcur = bnext;
        }
    }
    __syncthreads();
    // epilogue: bias + sigmoid -> sH (bf16)
#pragma unroll
    for (int t = 0; t < 12; ++t) {
        int n = (nh * 12 + t) * 16 + lm;
#pragma unroll
        for (int r = 0; r < 8; ++r) {
            int row = rt * 16 + lhi * 8 + r;
            sH[row * H_DIM + n] = f2bf(sigm(acc1[t][r] + sB1[n]));
        }
    }
    __syncthreads();

    // -------- Stage 2: latent = sigmoid(h @ W2 + b2), WMMA bf16 -----------
    v8f acc2[2];
    acc2[0] = zero8(); acc2[1] = zero8();
    const int m2 = rt * 16 + lm;
#pragma unroll
    for (int kk = 0; kk < 12; ++kk) {                  // K = 384 in 12 x 32
        int k0 = kk * 32 + lhi * 16;
        v8bf alo = *(const v8bf*)&sH[m2 * H_DIM + k0];
        v8bf ahi = *(const v8bf*)&sH[m2 * H_DIM + k0 + 8];
        v16bf a = cat8(alo, ahi);
#pragma unroll
        for (int t = 0; t < 2; ++t) {
            int n = (nh * 2 + t) * 16 + lm;
            v16bf b = *(const v16bf*)&sW2t[n * H_DIM + kk * 32 + lhi * 16];
            acc2[t] = __builtin_amdgcn_wmma_f32_16x16x32_bf16(
                false, a, false, b, (short)0, acc2[t], false, false);
        }
    }
#pragma unroll
    for (int t = 0; t < 2; ++t) {
        int n = (nh * 2 + t) * 16 + lm;
#pragma unroll
        for (int r = 0; r < 8; ++r) {
            int row = rt * 16 + lhi * 8 + r;
            sLat[row * L_DIM + n] = sigm(acc2[t][r] + sB2[n]);
        }
    }
    __syncthreads();

    // -------- Stage 3a: soft decision-tree routing -> mu (bf16 in LDS) ----
    {
        const int row = tid & 63;
        const int p   = tid >> 6;                      // wave-uniform (0..3)
        const float* lat = &sLat[row * L_DIM];
#pragma unroll
        for (int ei = 0; ei < 2; ++ei) {
            if (ei == 1 && p != 0) break;              // p==0 also does e=4
            const int e = (ei == 0) ? p : 4;
            const int* ph = &sPhi[e * 64];
            for (int l = 0; l < 64; ++l) {
                float m = 1.0f;
#pragma unroll
                for (int lev = 0; lev < 6; ++lev) {
                    int node = (1 << lev) + (l >> (6 - lev));
                    float dn = lat[ph[node]];
                    m *= ((l >> (5 - lev)) & 1) ? (1.0f - dn) : dn;
                }
                sMu[row * K_MU + e * 64 + l] = f2bf(m);
            }
        }
    }
    __syncthreads();

    // -------- Stage 3b: out = (mu @ pi) / E, WMMA bf16, K = 320 -----------
    if (wave < 4) {
        v8f acc3 = zero8();
        const int m3 = wave * 16 + lm;
#pragma unroll
        for (int kk = 0; kk < 10; ++kk) {
            int k0 = kk * 32 + lhi * 16;
            v8bf alo = *(const v8bf*)&sMu[m3 * K_MU + k0];
            v8bf ahi = *(const v8bf*)&sMu[m3 * K_MU + k0 + 8];
            v16bf a = cat8(alo, ahi);
            v16bf b = *(const v16bf*)&sPiT[lm * K_MU + kk * 32 + lhi * 16];
            acc3 = __builtin_amdgcn_wmma_f32_16x16x32_bf16(
                false, a, false, b, (short)0, acc3, false, false);
        }
#pragma unroll
        for (int r = 0; r < 8; ++r) {
            int row = rowBase + wave * 16 + lhi * 8 + r;
            out[row * O_DIM + lm] = acc3[r] * 0.2f;    // / E_NUM
        }
    }
}

extern "C" void kernel_launch(void* const* d_in, const int* in_sizes, int n_in,
                              void* d_out, int out_size, void* d_ws, size_t ws_size,
                              hipStream_t stream) {
    const float* X   = (const float*)d_in[0];
    const float* W1  = (const float*)d_in[1];
    const float* b1  = (const float*)d_in[2];
    const float* W2  = (const float*)d_in[3];
    const float* b2  = (const float*)d_in[4];
    const int*   phi = (const int*)  d_in[5];
    const float* pi  = (const float*)d_in[6];
    float* out = (float*)d_out;

    const int nRows = in_sizes[0] / F_DIM;             // 131072
    const int grid  = nRows / MT;                      // 2048 workgroups

    hipFuncSetAttribute((const void*)ldlf_fused_kernel,
                        hipFuncAttributeMaxDynamicSharedMemorySize, SMEM_BYTES);
    ldlf_fused_kernel<<<grid, THREADS, SMEM_BYTES, stream>>>(
        X, W1, b1, W2, b2, phi, pi, out);
}
